// ReverseLSTMLayer_695784702522
// MI455X (gfx1250) — matrix operand order repaired and checked by reading, not compiled, not run on hardware
//
#include <hip/hip_runtime.h>
#include <hip/hip_bf16.h>

// ---------------------------------------------------------------------------
// Reverse time-decay LSTM for MI455X (gfx1250, wave32, WMMA bf16).
//   S=2048, B=64, D=128, H=128, gates NG=512.
//
// Phase 0: convert W_ih / W_hh f32 -> bf16 *pre-packed in WMMA B-fragment
//          order* so GEMM kernels load fragments as contiguous b128.
// Phase 1: gates_pre[s,b,:] = x[s,b,:] @ W_ih^T + bias  (bf16 WMMA GEMM,
//          8192 WGs, coalesced float4 row stores via LDS bounce).
// Phase 2: sequential scan s = S-1..0, 4 WGs x 16 batch rows, W_hh fragments
//          held in registers across all 2048 steps; one-step-ahead
//          global_prefetch of the next gates_pre block hides HBM latency on
//          the serial critical path; NT hints for the streaming traffic.
// ---------------------------------------------------------------------------

#define S_LEN 2048
#define BATCH 64
#define DIN   128
#define HID   128
#define NG    512           // 4*H
#define GSTR  520           // padded LDS gates row stride (floats), %4==0

typedef __attribute__((ext_vector_type(16))) __bf16 v16bf;
typedef __attribute__((ext_vector_type(8)))  float  v8f;
typedef __attribute__((ext_vector_type(4)))  float  f4v;

union Frag {
    v16bf        bf;
    unsigned int u[8];
};

__device__ __forceinline__ unsigned int f2bf(float f) {
    union { float f; unsigned int u; } v;
    v.f = f;
    unsigned int u = v.u;
    unsigned int r = u + 0x7FFFu + ((u >> 16) & 1u);   // round to nearest even
    return r >> 16;
}

__device__ __forceinline__ float sigmoidf_fast(float x) {
    return 1.0f / (1.0f + __expf(-x));
}

// Load one pre-packed B fragment (8 dwords, contiguous per lane).
__device__ __forceinline__ void load_packed_frag(Frag& F,
                                                 const unsigned int* __restrict__ base) {
    const uint4* p = (const uint4*)base;
    uint4 a = p[0];
    uint4 b = p[1];
    F.u[0] = a.x; F.u[1] = a.y; F.u[2] = a.z; F.u[3] = a.w;
    F.u[4] = b.x; F.u[5] = b.y; F.u[6] = b.z; F.u[7] = b.w;
}

// ---------------------------------------------------------------------------
// Kernel 0: convert W (512x128 f32, row-major) into packed bf16 B-fragments.
//   Packed index: dword = ((nt*4 + kc)*32 + lane)*8 + v
//     nt = N-tile (0..31), kc = K-chunk (0..3), lane: N = nt*16 + (lane&15),
//     v  = fragment VGPR; halves hold K = 8*(lane>>4) + 2v (+8 if v>=4), +1.
//   Two matrices back-to-back: i >> 15 selects W_ih (0) or W_hh (1).
// ---------------------------------------------------------------------------
__global__ void cvt_pack_weights_kernel(const float* __restrict__ Wih,
                                        const float* __restrict__ Whh,
                                        unsigned int* __restrict__ dst) {
    int i = blockIdx.x * blockDim.x + threadIdx.x;   // 0 .. 65535
    if (i >= 2 * 32 * 4 * 32 * 8) return;
    int v    = i & 7;
    int lane = (i >> 3) & 31;
    int kc   = (i >> 8) & 3;
    int nt   = (i >> 10) & 31;
    int mat  = i >> 15;
    const float* W = mat ? Whh : Wih;
    int n = nt * 16 + (lane & 15);
    int k = kc * 32 + 8 * (lane >> 4) + 2 * v + ((v >> 2) << 3);
    unsigned lo = f2bf(W[n * DIN + k]);
    unsigned hi = f2bf(W[n * DIN + k + 1]);
    dst[i] = lo | (hi << 16);
}

// ---------------------------------------------------------------------------
// Kernel 1: gates_pre = x @ W_ih^T + bias.
//   x viewed as (S*B, 128) f32, gates_pre is (S*B, 512) f32.
//   One WG (512 thr / 16 waves) per 16-row M tile; each wave: 2 N-tiles.
// ---------------------------------------------------------------------------
__global__ __launch_bounds__(512)
void gemm_pre_kernel(const float* __restrict__ x,
                     const unsigned int* __restrict__ Wih_pk,
                     const float* __restrict__ bias,
                     float* __restrict__ gates_pre) {
    __shared__ unsigned int xLDS[16 * 64];     // 16 rows x 128 bf16
    __shared__ float        gLDS[16 * GSTR];   // staged result for coalesced store

    const int tid  = threadIdx.x;
    const int lane = tid & 31;
    const int wave = tid >> 5;
    const size_t m0 = (size_t)blockIdx.x * 16;

    // Stage x tile into LDS as packed bf16.
    for (int i = tid; i < 16 * 64; i += 512) {
        int row = i >> 6, dw = i & 63;
        float f0 = x[(m0 + row) * DIN + dw * 2];
        float f1 = x[(m0 + row) * DIN + dw * 2 + 1];
        xLDS[i] = f2bf(f0) | (f2bf(f1) << 16);
    }
    __syncthreads();

    const int nb0  = wave * 32;          // this wave's first N-tile column
    const int nt0  = wave * 2;           // this wave's first N-tile index
    const int r16  = lane & 15;
    const int hi16 = lane >> 4;

    v8f acc0 = {};
    v8f acc1 = {};

    #pragma unroll
    for (int kc = 0; kc < 4; ++kc) {
        // A fragment (16x32 bf16) from LDS.
        Frag A;
        int dwbase = r16 * 64 + kc * 16 + hi16 * 4;
        #pragma unroll
        for (int v = 0; v < 8; ++v) {
            int dw = (v < 4) ? v : (v + 4);
            A.u[v] = xLDS[dwbase + dw];
        }
        // Pre-packed B fragments: one b128 pair per lane.
        Frag B0, B1;
        load_packed_frag(B0, Wih_pk + (((nt0 + 0) * 4 + kc) * 32 + lane) * 8);
        load_packed_frag(B1, Wih_pk + (((nt0 + 1) * 4 + kc) * 32 + lane) * 8);
        acc0 = __builtin_amdgcn_wmma_f32_16x16x32_bf16(
                   false, A.bf, false, B0.bf, (short)0, acc0, false, false);
        acc1 = __builtin_amdgcn_wmma_f32_16x16x32_bf16(
                   false, A.bf, false, B1.bf, (short)0, acc1, false, false);
    }

    // Stage C tiles in LDS (C/D layout: vgpr r -> M=r / r+8, N = lane&15).
    {
        int rowbase = hi16 * 8;
        int col0 = nb0 + r16;
        #pragma unroll
        for (int r = 0; r < 8; ++r) {
            gLDS[(rowbase + r) * GSTR + col0]      = acc0[r];
            gLDS[(rowbase + r) * GSTR + col0 + 16] = acc1[r];
        }
    }
    __syncthreads();

    // Coalesced write: wave w streams row w (512 floats) as float4s + bias.
    {
        const f4v* brow = (const f4v*)bias;
        f4v* grow = (f4v*)(gates_pre + (m0 + wave) * NG);
        #pragma unroll
        for (int q = lane; q < NG / 4; q += 32) {
            const f4v a = *(const f4v*)&gLDS[wave * GSTR + q * 4];
            grow[q] = a + brow[q];
        }
    }
}

// ---------------------------------------------------------------------------
// Kernel 2: the reverse scan.  grid = 4 WGs (16 batch rows each), 512 threads.
// ---------------------------------------------------------------------------
__global__ __launch_bounds__(512)
void scan_kernel(const float* __restrict__ gates_pre,
                 const float* __restrict__ timev,
                 const unsigned int* __restrict__ Whh_pk,
                 const float* __restrict__ w_t,
                 const float* __restrict__ b_t,
                 float* __restrict__ out) {
    __shared__ float        gatesLDS[16 * GSTR];   // 16 x 512 (+pad) f32
    __shared__ unsigned int hLDS[16 * 64];         // 16 x 128 bf16

    const int tid  = threadIdx.x;
    const int lane = tid & 31;
    const int wave = tid >> 5;
    const int b0   = blockIdx.x * 16;

    // h starts at zero.
    for (int i = tid; i < 16 * 64; i += 512) hLDS[i] = 0u;

    // Warm the first step's 32 KB gates_pre block (16 contiguous rows):
    // 512 threads x 64 B = 32 KB.
    {
        const char* pf0 = (const char*)(gates_pre +
                          ((size_t)(S_LEN - 1) * BATCH + b0) * NG);
        __builtin_prefetch(pf0 + tid * 64, 0, 1);
    }

    // Preload this wave's W_hh B-fragments (2 N-tiles x 4 K-chunks, 64 VGPRs),
    // loop-invariant across all 2048 steps.  Pre-packed: pure b128 loads.
    const int nb0 = wave * 32;
    const int nt0 = wave * 2;
    Frag Bf[2][4];
    #pragma unroll
    for (int t = 0; t < 2; ++t) {
        #pragma unroll
        for (int kc = 0; kc < 4; ++kc) {
            load_packed_frag(Bf[t][kc],
                             Whh_pk + (((nt0 + t) * 4 + kc) * 32 + lane) * 8);
        }
    }

    // Elementwise role: each thread owns 4 contiguous h positions of one row.
    const int eb = tid >> 5;          // batch row in tile (0..15)
    const int eh = (tid & 31) * 4;    // h base (0..124)
    f4v wt = *(const f4v*)&w_t[eh];
    f4v bt = *(const f4v*)&b_t[eh];
    float c[4] = {0.0f, 0.0f, 0.0f, 0.0f};

    const int r16  = lane & 15;
    const int hi16 = lane >> 4;

    for (int s = S_LEN - 1; s >= 0; --s) {
        __syncthreads();   // hLDS (from previous step) visible to all waves

        // Prefetch next step's contiguous 32 KB gates_pre block; the whole
        // WMMA + elementwise phase below hides the HBM latency.
        {
            int sn = (s > 0) ? (s - 1) : 0;
            const char* pfn = (const char*)(gates_pre +
                              ((size_t)sn * BATCH + b0) * NG);
            __builtin_prefetch(pfn + tid * 64, 0, 1);
        }

        // ---- recurrent GEMM: gates_rec = h @ W_hh^T ------------------------
        v8f acc0 = {};
        v8f acc1 = {};
        #pragma unroll
        for (int kc = 0; kc < 4; ++kc) {
            Frag A;
            int dwbase = r16 * 64 + kc * 16 + hi16 * 4;
            #pragma unroll
            for (int v = 0; v < 8; ++v) {
                int dw = (v < 4) ? v : (v + 4);
                A.u[v] = hLDS[dwbase + dw];
            }
            acc0 = __builtin_amdgcn_wmma_f32_16x16x32_bf16(
                       false, A.bf, false, Bf[0][kc].bf, (short)0, acc0, false, false);
            acc1 = __builtin_amdgcn_wmma_f32_16x16x32_bf16(
                       false, A.bf, false, Bf[1][kc].bf, (short)0, acc1, false, false);
        }
        {
            int rowbase = hi16 * 8;
            int col0 = nb0 + r16;
            #pragma unroll
            for (int r = 0; r < 8; ++r) {
                gatesLDS[(rowbase + r) * GSTR + col0]      = acc0[r];
                gatesLDS[(rowbase + r) * GSTR + col0 + 16] = acc1[r];
            }
        }
        __syncthreads();   // gates staged; all hLDS reads done

        // ---- elementwise LSTM update (float4, NT streaming loads) ----------
        const size_t rowidx = (size_t)s * BATCH + (b0 + eb);
        const float  tval   = timev[rowidx];
        const float* gp     = gates_pre + rowidx * NG;

        f4v gi4 = __builtin_nontemporal_load((const f4v*)&gp[eh]);
        f4v gf4 = __builtin_nontemporal_load((const f4v*)&gp[eh + 128]);
        f4v gg4 = __builtin_nontemporal_load((const f4v*)&gp[eh + 256]);
        f4v go4 = __builtin_nontemporal_load((const f4v*)&gp[eh + 384]);
        gi4 += *(const f4v*)&gatesLDS[eb * GSTR + eh];
        gf4 += *(const f4v*)&gatesLDS[eb * GSTR + eh + 128];
        gg4 += *(const f4v*)&gatesLDS[eb * GSTR + eh + 256];
        go4 += *(const f4v*)&gatesLDS[eb * GSTR + eh + 384];

        f4v hv;
        #pragma unroll
        for (int j = 0; j < 4; ++j) {
            float i_ = sigmoidf_fast(gi4[j]);
            float f_ = sigmoidf_fast(gf4[j]);
            float g_ = tanhf(gg4[j]);
            float o_ = sigmoidf_fast(go4[j]);
            float dz    = tval * wt[j] + bt[j];
            float decay = __expf(-fmaxf(dz, 0.0f));
            c[j]  = f_ * c[j] * decay + i_ * g_;
            hv[j] = o_ * tanhf(c[j]);
        }

        __builtin_nontemporal_store(hv, (f4v*)&out[rowidx * HID + eh]);

        // repack h as bf16 for next step's A fragments
        hLDS[eb * 64 + (eh >> 1)]     = f2bf(hv[0]) | (f2bf(hv[1]) << 16);
        hLDS[eb * 64 + (eh >> 1) + 1] = f2bf(hv[2]) | (f2bf(hv[3]) << 16);

        if (s == 0) {   // final (h, c) appended after outputs
            size_t fin = (size_t)S_LEN * BATCH * HID;
            f4v cv; cv[0] = c[0]; cv[1] = c[1]; cv[2] = c[2]; cv[3] = c[3];
            *(f4v*)&out[fin + (size_t)(b0 + eb) * HID + eh] = hv;
            *(f4v*)&out[fin + (size_t)BATCH * HID + (size_t)(b0 + eb) * HID + eh] = cv;
        }
    }
}

// ---------------------------------------------------------------------------
extern "C" void kernel_launch(void* const* d_in, const int* in_sizes, int n_in,
                              void* d_out, int out_size, void* d_ws, size_t ws_size,
                              hipStream_t stream) {
    const float* x    = (const float*)d_in[0];   // (S,B,D)
    const float* tv   = (const float*)d_in[1];   // (S,B,1)
    const float* Wih  = (const float*)d_in[2];   // (512,128)
    const float* Whh  = (const float*)d_in[3];   // (512,128)
    const float* bias = (const float*)d_in[4];   // (512,)
    const float* w_t  = (const float*)d_in[5];   // (128,)
    const float* b_t  = (const float*)d_in[6];   // (128,)
    float* out = (float*)d_out;

    // Workspace layout:
    //   [0, 128KB)        W_ih packed bf16 fragments (32768 dwords)
    //   [128KB, 256KB)    W_hh packed bf16 fragments
    //   [256KB, +256MB)   gates_pre f32 (S*B, 512)
    unsigned int* Wih_pk = (unsigned int*)d_ws;
    unsigned int* Whh_pk = Wih_pk + 32 * 4 * 32 * 8;
    float* gates_pre = (float*)((char*)d_ws + (size_t)256 * 1024);

    cvt_pack_weights_kernel<<<(2 * 32 * 4 * 32 * 8 + 255) / 256, 256, 0, stream>>>(
        Wih, Whh, Wih_pk);

    gemm_pre_kernel<<<(S_LEN * BATCH) / 16, 512, 0, stream>>>(x, Wih_pk, bias, gates_pre);

    scan_kernel<<<BATCH / 16, 512, 0, stream>>>(gates_pre, tv, Whh_pk, w_t, b_t, out);
}
